// ImageResampler_1400159338513
// MI455X (gfx1250) — matrix operand order repaired and checked by reading, not compiled, not run on hardware
//
#include <hip/hip_runtime.h>
#include <cstdint>
#include <cstddef>

typedef __attribute__((ext_vector_type(16))) _Float16 v16h;
typedef __attribute__((ext_vector_type(8)))  float    v8f;
typedef __attribute__((ext_vector_type(4)))  int      v4i;

#define DEPTH_N 6
#define BATCH   16
#define NLAT    64
#define DIM     768
#define INNER   1024
#define FFD     3072

#if defined(__gfx1250__) && __has_builtin(__builtin_amdgcn_global_load_async_to_lds_b128)
#define ASYNC_LDS 1
#endif

#ifdef ASYNC_LDS
__device__ __forceinline__ void async_copy16(const void* g, void* l) {
  // global -> LDS direct copy, 16B per lane, tracked by ASYNCcnt
  __builtin_amdgcn_global_load_async_to_lds_b128(
      (__attribute__((address_space(1))) v4i*)(v4i*)const_cast<void*>(g),
      (__attribute__((address_space(3))) v4i*)(v4i*)l, 0, 0);
}
__device__ __forceinline__ void wait_async0() {
#if __has_builtin(__builtin_amdgcn_s_wait_asynccnt)
  __builtin_amdgcn_s_wait_asynccnt(0);
#else
  asm volatile("s_wait_asynccnt 0x0" ::: "memory");
#endif
}
#endif

// ---------------------------------------------------------------------------
// WMMA fragment loaders (layouts per CDNA5 ISA §7.12.2, wave32)
// A: 16x32 f16, lane L: row = L&15; K = (L>>4)*8 + 2p (p<4), 16+(L>>4)*8+2(p-4)
// B: 32x16 f16 from transposed LDS [col][K]: lane L: col = L&15; K = (L>>4)*8 + 2j (j<4), 16+(L>>4)*8+2(j-4)
// ---------------------------------------------------------------------------
__device__ __forceinline__ v16h load_a_frag(const _Float16* base, int ld, int row0, int k0) {
  union { v16h v; unsigned u[8]; } f;
  int lane = threadIdx.x & 31;
  int r = lane & 15, hi = lane >> 4;
  const _Float16* rp = base + (size_t)(row0 + r) * ld + k0;
#pragma unroll
  for (int p = 0; p < 8; ++p) {
    int k = (p < 4) ? (hi * 8 + 2 * p) : (16 + hi * 8 + 2 * (p - 4));
    f.u[p] = *(const unsigned*)(rp + k);
  }
  return f.v;
}

__device__ __forceinline__ v16h load_b_frag(const _Float16* baseT, int ld, int col0, int k0) {
  union { v16h v; unsigned u[8]; } f;
  int lane = threadIdx.x & 31;
  int n = lane & 15, hi = lane >> 4;
  const _Float16* cp = baseT + (size_t)(col0 + n) * ld + k0;
#pragma unroll
  for (int j = 0; j < 8; ++j) f.u[j] = *(const unsigned*)(cp + ((j < 4) ? (hi * 8 + 2 * j) : (16 + hi * 8 + 2 * (j - 4))));
  return f.v;
}

// ---------------------------------------------------------------------------
// Generic WMMA GEMM: C[M,N] = A[M,K](f16,row-major) @ B[K,N](f16,row-major)
// Optional: +bias(f32), exact GELU, f32 residual-accumulate, f16/f32 output.
// BM=128 BN=128 BK=32, 256 threads = 8 waves (2x4), wave tile 64x32
// -> 8 v_wmma per K-step from 4 A-frags + 2 B-frags.
// ---------------------------------------------------------------------------
#define GBM 128
#define GBN 128
#define GBK 32
#define GLDA 40   // halves: 80B row stride (16B aligned)
#define GLDB 40

template<bool BIAS, bool GELU, bool RESID, bool OUT16>
__launch_bounds__(256)
__global__ void gemm_kernel(const _Float16* __restrict__ A, const _Float16* __restrict__ B,
                            const float* __restrict__ bias,
                            float* __restrict__ outF, _Float16* __restrict__ outH,
                            int M, int N, int K) {
  __shared__ _Float16 As[GBM * GLDA];
  __shared__ _Float16 Bt[GBN * GLDB];
  const int tid = threadIdx.x;
  const int wave = tid >> 5;
  const int wm = wave >> 2, wn = wave & 3;       // 2 x 4 wave grid
  const int m0 = blockIdx.y * GBM, n0 = blockIdx.x * GBN;
  v8f acc[4][2] = {};
  for (int k0 = 0; k0 < K; k0 += GBK) {
    __syncthreads();
    // ---- A tile: 128 rows x 32 halves, 16B per lane-op, 2 ops/thread ----
#pragma unroll
    for (int i = 0; i < 2; ++i) {
      int idx = tid + i * 256;                   // 0..511
      int r = idx >> 2, c = (idx & 3) * 8;       // 4x b128 per row
      int gr = m0 + r;
#ifdef ASYNC_LDS
      if (gr < M) async_copy16(A + (size_t)gr * K + k0 + c, As + r * GLDA + c);
#else
      uint4 val = {0u, 0u, 0u, 0u};
      if (gr < M) val = *(const uint4*)(A + (size_t)gr * K + k0 + c);
      *(uint4*)(As + r * GLDA + c) = val;
#endif
    }
    // ---- B tile: 32 rows x 128 cols, packed transpose into Bt[col][K] ----
#pragma unroll
    for (int i = 0; i < 2; ++i) {
      int idx = tid + i * 256;                   // 0..511, each: 4 K-rows x 2 cols
      int kg = idx >> 6;                         // 0..7 -> K rows 4*kg..4*kg+3
      int nc = (idx & 63) * 2;                   // col pair
      const _Float16* bp = B + (size_t)(k0 + kg * 4) * N + n0 + nc;
      unsigned u0 = *(const unsigned*)(bp);
      unsigned u1 = *(const unsigned*)(bp + N);
      unsigned u2 = *(const unsigned*)(bp + 2 * N);
      unsigned u3 = *(const unsigned*)(bp + 3 * N);
      uint2 t0, t1;
      t0.x = (u0 & 0xffffu) | (u1 << 16);        // col nc   : K=4kg..4kg+1
      t0.y = (u2 & 0xffffu) | (u3 << 16);        //            K=4kg+2..4kg+3
      t1.x = (u0 >> 16) | (u1 & 0xffff0000u);    // col nc+1
      t1.y = (u2 >> 16) | (u3 & 0xffff0000u);
      *(uint2*)(Bt + (nc + 0) * GLDB + kg * 4) = t0;
      *(uint2*)(Bt + (nc + 1) * GLDB + kg * 4) = t1;
      // prefetch next K tile of B (lowers to global_prefetch_b8)
      if (k0 + GBK < K) __builtin_prefetch(bp + (size_t)GBK * N, 0, 1);
    }
    if (k0 + GBK < K) {
      int r = tid >> 1, c = (tid & 1) * 16;
      int gr = m0 + r;
      if (gr < M) __builtin_prefetch(A + (size_t)gr * K + k0 + GBK + c, 0, 1);
    }
#ifdef ASYNC_LDS
    wait_async0();
#endif
    __syncthreads();
    v16h af[4], bf[2];
#pragma unroll
    for (int mt = 0; mt < 4; ++mt) af[mt] = load_a_frag(As, GLDA, wm * 64 + mt * 16, 0);
#pragma unroll
    for (int nt = 0; nt < 2; ++nt) bf[nt] = load_b_frag(Bt, GLDB, wn * 32 + nt * 16, 0);
#pragma unroll
    for (int mt = 0; mt < 4; ++mt)
#pragma unroll
      for (int nt = 0; nt < 2; ++nt)
        acc[mt][nt] = __builtin_amdgcn_wmma_f32_16x16x32_f16(
            false, af[mt], false, bf[nt], (short)0, acc[mt][nt], false, false);
  }
  const int lane = tid & 31, cn = lane & 15, hi = lane >> 4;
#pragma unroll
  for (int mt = 0; mt < 4; ++mt)
#pragma unroll
    for (int nt = 0; nt < 2; ++nt) {
      int col = n0 + wn * 32 + nt * 16 + cn;
      float bv = 0.f;
      if (BIAS) bv = bias[col];
#pragma unroll
      for (int v = 0; v < 8; ++v) {
        int row = m0 + wm * 64 + mt * 16 + hi * 8 + v;
        if (row < M) {
          float x = acc[mt][nt][v] + bv;
          if (GELU) x = 0.5f * x * (1.f + erff(x * 0.70710678f));
          size_t off = (size_t)row * N + col;
          if (RESID) x += outF[off];
          if (OUT16) outH[off] = (_Float16)x; else outF[off] = x;
        }
      }
    }
}

// ---------------------------------------------------------------------------
// LayerNorm over last dim C (block per row), f16 or f32 output
// ---------------------------------------------------------------------------
template<bool OUT16>
__launch_bounds__(256)
__global__ void ln_kernel(const float* __restrict__ x, const float* __restrict__ w,
                          const float* __restrict__ b,
                          float* __restrict__ outF, _Float16* __restrict__ outH, int C) {
  __shared__ float red[256];
  __shared__ float stat[2];
  const int row = blockIdx.x;
  const int tid = threadIdx.x;
  const float* xr = x + (size_t)row * C;
  float s = 0.f, s2 = 0.f;
  for (int c = tid; c < C; c += 256) { float v = xr[c]; s += v; s2 += v * v; }
  red[tid] = s; __syncthreads();
  for (int o = 128; o > 0; o >>= 1) { if (tid < o) red[tid] += red[tid + o]; __syncthreads(); }
  if (tid == 0) stat[0] = red[0] / C;
  __syncthreads();
  red[tid] = s2; __syncthreads();
  for (int o = 128; o > 0; o >>= 1) { if (tid < o) red[tid] += red[tid + o]; __syncthreads(); }
  if (tid == 0) { float m = stat[0]; stat[1] = rsqrtf(red[0] / C - m * m + 1e-5f); }
  __syncthreads();
  float mean = stat[0], inv = stat[1];
  for (int c = tid; c < C; c += 256) {
    float v = (xr[c] - mean) * inv * w[c] + b[c];
    size_t off = (size_t)row * C + c;
    if (OUT16) outH[off] = (_Float16)v; else outF[off] = v;
  }
}

__global__ void cvt_kernel(const float* __restrict__ in, _Float16* __restrict__ out, long long n) {
  long long i = (long long)blockIdx.x * 256 + threadIdx.x;
  if (i < n) out[i] = (_Float16)in[i];
}

// ---------------------------------------------------------------------------
// Flash attention per (batch, head): 64 queries, keys streamed in 64-chunks.
// Q[B*64, 1024] f16 ; KVx[B*n1, 2048] ; KVl[B*64, 2048] (k cols 0..1023, v 1024..2047)
// 128 threads = 4 waves; wave owns 16 query rows. Net scale qk/8 (double-sqrt).
// ---------------------------------------------------------------------------
__launch_bounds__(128)
__global__ void attn_kernel(const _Float16* __restrict__ Q, const _Float16* __restrict__ KVx,
                            const _Float16* __restrict__ KVl, _Float16* __restrict__ O, int n1) {
  const int h = blockIdx.x, b = blockIdx.y;
  const int Lk = n1 + NLAT;
  __shared__ _Float16 Qs[64 * 72];
  __shared__ _Float16 Ks[64 * 72];   // [key][dh]  (== B-frag transposed layout for S)
  __shared__ _Float16 Vt[64 * 72];   // [dh][key]  (B-frag transposed layout for P@V)
  __shared__ _Float16 Ps[64 * 72];   // [query][key-in-chunk]
  const int tid = threadIdx.x, wave = tid >> 5, lane = tid & 31, ln = lane & 15, hi = lane >> 4;
#pragma unroll
  for (int i = 0; i < 16; ++i) {
    int idx = tid + i * 128;
    int r = idx >> 5, c = (idx & 31) * 2;
    *(unsigned*)(Qs + r * 72 + c) =
        *(const unsigned*)(Q + ((size_t)(b * NLAT + r)) * INNER + h * 64 + c);
  }
  float m[8], l[8];
  v8f oacc[4] = {};
#pragma unroll
  for (int v = 0; v < 8; ++v) { m[v] = -1e30f; l[v] = 0.f; }
  const int qb = wave * 16;
  for (int kc = 0; kc < Lk; kc += 64) {
    __syncthreads();
#pragma unroll
    for (int i = 0; i < 16; ++i) {
      int idx = tid + i * 128;
      int r = idx >> 5, c = (idx & 31) * 2;
      int kg = kc + r;
      unsigned uk = 0, uv = 0;
      if (kg < n1) {
        const _Float16* src = KVx + ((size_t)b * n1 + kg) * (2 * INNER) + h * 64;
        uk = *(const unsigned*)(src + c);
        uv = *(const unsigned*)(src + INNER + c);
      } else if (kg < Lk) {
        const _Float16* src = KVl + ((size_t)b * NLAT + (kg - n1)) * (2 * INNER) + h * 64;
        uk = *(const unsigned*)(src + c);
        uv = *(const unsigned*)(src + INNER + c);
      }
      *(unsigned*)(Ks + r * 72 + c) = uk;
      Vt[(c + 0) * 72 + r] = ((const _Float16*)&uv)[0];
      Vt[(c + 1) * 72 + r] = ((const _Float16*)&uv)[1];
    }
    __syncthreads();
    // S = Q @ K^T for this wave's 16 queries x 64 keys
    v8f sacc[4] = {};
#pragma unroll
    for (int ks = 0; ks < 64; ks += 32) {
      v16h a = load_a_frag(Qs, 72, qb, ks);
#pragma unroll
      for (int nt = 0; nt < 4; ++nt) {
        v16h bf = load_b_frag(Ks, 72, nt * 16, ks);
        sacc[nt] = __builtin_amdgcn_wmma_f32_16x16x32_f16(
            false, a, false, bf, (short)0, sacc[nt], false, false);
      }
    }
    // scale + tail mask + row max (D layout: row = v + hi*8, col = ln across 16 lanes)
    float rm[8];
#pragma unroll
    for (int v = 0; v < 8; ++v) rm[v] = -1e30f;
#pragma unroll
    for (int nt = 0; nt < 4; ++nt) {
      bool valid = (kc + nt * 16 + ln) < Lk;
#pragma unroll
      for (int v = 0; v < 8; ++v) {
        float x = valid ? sacc[nt][v] * 0.125f : -1e30f;
        sacc[nt][v] = x;
        rm[v] = fmaxf(rm[v], x);
      }
    }
#pragma unroll
    for (int o = 1; o < 16; o <<= 1)
#pragma unroll
      for (int v = 0; v < 8; ++v) rm[v] = fmaxf(rm[v], __shfl_xor(rm[v], o, 32));
    float alpha[8], rs[8];
#pragma unroll
    for (int v = 0; v < 8; ++v) {
      float mn = fmaxf(m[v], rm[v]);
      alpha[v] = __expf(m[v] - mn);
      m[v] = mn; rs[v] = 0.f;
    }
#pragma unroll
    for (int nt = 0; nt < 4; ++nt)
#pragma unroll
      for (int v = 0; v < 8; ++v) {
        float p = __expf(sacc[nt][v] - m[v]);
        sacc[nt][v] = p; rs[v] += p;
      }
#pragma unroll
    for (int o = 1; o < 16; o <<= 1)
#pragma unroll
      for (int v = 0; v < 8; ++v) rs[v] += __shfl_xor(rs[v], o, 32);
#pragma unroll
    for (int v = 0; v < 8; ++v) l[v] = l[v] * alpha[v] + rs[v];
    // stage P to LDS (A-frag layout needs K packed per lane), rescale O
#pragma unroll
    for (int nt = 0; nt < 4; ++nt)
#pragma unroll
      for (int v = 0; v < 8; ++v)
        Ps[(qb + hi * 8 + v) * 72 + nt * 16 + ln] = (_Float16)sacc[nt][v];
#pragma unroll
    for (int nt = 0; nt < 4; ++nt)
#pragma unroll
      for (int v = 0; v < 8; ++v) oacc[nt][v] *= alpha[v];
    __syncthreads();
    // O += P @ V
#pragma unroll
    for (int ks = 0; ks < 64; ks += 32) {
      v16h a = load_a_frag(Ps, 72, qb, ks);
#pragma unroll
      for (int nt = 0; nt < 4; ++nt) {
        v16h bf = load_b_frag(Vt, 72, nt * 16, ks);
        oacc[nt] = __builtin_amdgcn_wmma_f32_16x16x32_f16(
            false, a, false, bf, (short)0, oacc[nt], false, false);
      }
    }
  }
#pragma unroll
  for (int nt = 0; nt < 4; ++nt)
#pragma unroll
    for (int v = 0; v < 8; ++v) {
      int row = b * NLAT + qb + hi * 8 + v;
      int col = h * 64 + nt * 16 + ln;
      O[(size_t)row * INNER + col] = (_Float16)(oacc[nt][v] / l[v]);
    }
}

// ---------------------------------------------------------------------------
// Host orchestration
// ---------------------------------------------------------------------------
extern "C" void kernel_launch(void* const* d_in, const int* in_sizes, int n_in,
                              void* d_out, int out_size, void* d_ws, size_t ws_size,
                              hipStream_t stream) {
  (void)in_sizes; (void)n_in; (void)out_size; (void)ws_size;
  const float* id_emb = (const float*)d_in[0];
  // reference uses reversed order: feats[0] <- f3 (deepest)
  const float* fsrc[4] = {(const float*)d_in[4], (const float*)d_in[3],
                          (const float*)d_in[2], (const float*)d_in[1]};
  const float* pw[4] = {(const float*)d_in[11], (const float*)d_in[9],
                        (const float*)d_in[7],  (const float*)d_in[5]};
  const float* pb[4] = {(const float*)d_in[12], (const float*)d_in[10],
                        (const float*)d_in[8],  (const float*)d_in[6]};
  const float* ln1w = (const float*)d_in[13];
  const float* ln1b = (const float*)d_in[14];
  const float* ln2w = (const float*)d_in[15];
  const float* ln2b = (const float*)d_in[16];
  const float* wq   = (const float*)d_in[17];
  const float* wkv  = (const float*)d_in[18];
  const float* wo   = (const float*)d_in[19];
  const float* fflnw = (const float*)d_in[20];
  const float* fflnb = (const float*)d_in[21];
  const float* ffw1  = (const float*)d_in[22];
  const float* ffw2  = (const float*)d_in[23];
  const float* pjw = (const float*)d_in[24];
  const float* pjb = (const float*)d_in[25];
  const float* now = (const float*)d_in[26];
  const float* nob = (const float*)d_in[27];

  const int lens[4] = {49, 196, 784, 3136};
  const int frd[4]  = {512, 256, 128, 64};

  char* wp = (char*)d_ws;
  auto alloc = [&](size_t bytes) -> void* {
    void* r = (void*)wp; wp += (bytes + 255) & ~(size_t)255; return r;
  };

  _Float16* fh[4]; _Float16* pwh[4]; float* feat[4];
  for (int i = 0; i < 4; ++i) {
    fh[i]   = (_Float16*)alloc((size_t)BATCH * lens[i] * frd[i] * 2);
    pwh[i]  = (_Float16*)alloc((size_t)frd[i] * DIM * 2);
    feat[i] = (float*)alloc((size_t)BATCH * lens[i] * DIM * 4);
  }
  _Float16* wq_h  = (_Float16*)alloc((size_t)DEPTH_N * 4 * DIM * INNER * 2);
  _Float16* wkv_h = (_Float16*)alloc((size_t)DEPTH_N * 4 * DIM * 2 * INNER * 2);
  _Float16* wo_h  = (_Float16*)alloc((size_t)DEPTH_N * 4 * INNER * DIM * 2);
  _Float16* w1_h  = (_Float16*)alloc((size_t)DEPTH_N * DIM * FFD * 2);
  _Float16* w2_h  = (_Float16*)alloc((size_t)DEPTH_N * FFD * DIM * 2);
  _Float16* pj_h  = (_Float16*)alloc((size_t)DIM * DIM * 2);
  float*    lat   = (float*)alloc((size_t)BATCH * NLAT * DIM * 4);
  _Float16* xn    = (_Float16*)alloc((size_t)BATCH * 3136 * DIM * 2);
  _Float16* latn  = (_Float16*)alloc((size_t)BATCH * NLAT * DIM * 2);
  _Float16* qh    = (_Float16*)alloc((size_t)BATCH * NLAT * INNER * 2);
  _Float16* kvx   = (_Float16*)alloc((size_t)BATCH * 3136 * 2 * INNER * 2);
  _Float16* kvl   = (_Float16*)alloc((size_t)BATCH * NLAT * 2 * INNER * 2);
  _Float16* ao    = (_Float16*)alloc((size_t)BATCH * NLAT * INNER * 2);
  _Float16* ffh   = (_Float16*)alloc((size_t)BATCH * NLAT * FFD * 2);
  float*    tmp   = (float*)alloc((size_t)BATCH * NLAT * DIM * 4);

  auto cvt = [&](const float* src, _Float16* dst, long long n) {
    cvt_kernel<<<dim3((unsigned)((n + 255) / 256)), 256, 0, stream>>>(src, dst, n);
  };
  for (int i = 0; i < 4; ++i) {
    cvt(fsrc[i], fh[i], (long long)BATCH * lens[i] * frd[i]);
    cvt(pw[i], pwh[i], (long long)frd[i] * DIM);
  }
  cvt(wq,  wq_h,  (long long)DEPTH_N * 4 * DIM * INNER);
  cvt(wkv, wkv_h, (long long)DEPTH_N * 4 * DIM * 2 * INNER);
  cvt(wo,  wo_h,  (long long)DEPTH_N * 4 * INNER * DIM);
  cvt(ffw1, w1_h, (long long)DEPTH_N * DIM * FFD);
  cvt(ffw2, w2_h, (long long)DEPTH_N * FFD * DIM);
  cvt(pjw, pj_h,  (long long)DIM * DIM);

  // latents = id_embedding (f32 working copy)
  (void)hipMemcpyAsync(lat, id_emb, (size_t)BATCH * NLAT * DIM * 4,
                       hipMemcpyDeviceToDevice, stream);

  // per-scale input projections: feat[i] = f @ p_w + p_b
  for (int i = 0; i < 4; ++i) {
    int M = BATCH * lens[i], N = DIM, K = frd[i];
    dim3 g(N / GBN, (M + GBM - 1) / GBM);
    gemm_kernel<true, false, false, false><<<g, 256, 0, stream>>>(
        fh[i], pwh[i], pb[i], feat[i], nullptr, M, N, K);
  }

  for (int d = 0; d < DEPTH_N; ++d) {
    for (int i = 0; i < 4; ++i) {
      int idx = d * 4 + i;
      int n1 = lens[i];
      int Mx = BATCH * n1;
      ln_kernel<true><<<Mx, 256, 0, stream>>>(
          feat[i], ln1w + (size_t)idx * DIM, ln1b + (size_t)idx * DIM, nullptr, xn, DIM);
      ln_kernel<true><<<BATCH * NLAT, 256, 0, stream>>>(
          lat, ln2w + (size_t)idx * DIM, ln2b + (size_t)idx * DIM, nullptr, latn, DIM);
      {
        int M = BATCH * NLAT, N = INNER, K = DIM;
        dim3 g(N / GBN, (M + GBM - 1) / GBM);
        gemm_kernel<false, false, false, true><<<g, 256, 0, stream>>>(
            latn, wq_h + (size_t)idx * DIM * INNER, nullptr, nullptr, qh, M, N, K);
      }
      {
        int M = Mx, N = 2 * INNER, K = DIM;
        dim3 g(N / GBN, (M + GBM - 1) / GBM);
        gemm_kernel<false, false, false, true><<<g, 256, 0, stream>>>(
            xn, wkv_h + (size_t)idx * DIM * 2 * INNER, nullptr, nullptr, kvx, M, N, K);
      }
      {
        int M = BATCH * NLAT, N = 2 * INNER, K = DIM;
        dim3 g(N / GBN, (M + GBM - 1) / GBM);
        gemm_kernel<false, false, false, true><<<g, 256, 0, stream>>>(
            latn, wkv_h + (size_t)idx * DIM * 2 * INNER, nullptr, nullptr, kvl, M, N, K);
      }
      attn_kernel<<<dim3(16, BATCH), 128, 0, stream>>>(qh, kvx, kvl, ao, n1);
      {
        int M = BATCH * NLAT, N = DIM, K = INNER;
        dim3 g(N / GBN, (M + GBM - 1) / GBM);
        gemm_kernel<false, false, true, false><<<g, 256, 0, stream>>>(
            ao, wo_h + (size_t)idx * INNER * DIM, nullptr, lat, nullptr, M, N, K);
      }
    }
    // FeedForward: lat += gelu(LN(lat) @ w1) @ w2
    ln_kernel<true><<<BATCH * NLAT, 256, 0, stream>>>(
        lat, fflnw + (size_t)d * DIM, fflnb + (size_t)d * DIM, nullptr, latn, DIM);
    {
      int M = BATCH * NLAT, N = FFD, K = DIM;
      dim3 g(N / GBN, (M + GBM - 1) / GBM);
      gemm_kernel<false, true, false, true><<<g, 256, 0, stream>>>(
          latn, w1_h + (size_t)d * DIM * FFD, nullptr, nullptr, ffh, M, N, K);
    }
    {
      int M = BATCH * NLAT, N = DIM, K = FFD;
      dim3 g(N / GBN, (M + GBM - 1) / GBM);
      gemm_kernel<false, false, true, false><<<g, 256, 0, stream>>>(
          ffh, w2_h + (size_t)d * FFD * DIM, nullptr, lat, nullptr, M, N, K);
    }
  }
  // head: tmp = lat @ proj_w + proj_b ; out = LN(tmp)
  cvt(lat, latn, (long long)BATCH * NLAT * DIM);
  {
    int M = BATCH * NLAT, N = DIM, K = DIM;
    dim3 g(N / GBN, (M + GBM - 1) / GBM);
    gemm_kernel<true, false, false, false><<<g, 256, 0, stream>>>(
        latn, pj_h, pjb, tmp, nullptr, M, N, K);
  }
  ln_kernel<false><<<BATCH * NLAT, 256, 0, stream>>>(
      tmp, now, nob, (float*)d_out, nullptr, DIM);
}
